// TransformerEncoderLayer_47691316854955
// MI455X (gfx1250) — matrix operand order repaired and checked by reading, not compiled, not run on hardware
//
#include <hip/hip_runtime.h>
#include <math.h>

// ---------------- CDNA5 WMMA types ----------------
typedef __attribute__((ext_vector_type(16))) __bf16        v16bf;
typedef __attribute__((ext_vector_type(8)))  float         v8f;
typedef __attribute__((ext_vector_type(4)))  unsigned int  v4u;

union Frag {                 // 16 bf16 = 32 bytes (A or B fragment of 16x16x32 bf16 WMMA)
    v16bf           v;
    v4u             u[2];
    unsigned short  s[16];
};

union Pack8 {                // 8 bf16 = one b128 store
    v4u             u;
    unsigned short  s[8];
};

__device__ __forceinline__ unsigned short f2bf(float f) {
    unsigned u = __builtin_bit_cast(unsigned, f);
    return (unsigned short)((u + 0x7FFFu + ((u >> 16) & 1u)) >> 16);   // RNE
}

__device__ __forceinline__ v8f wmma_bf16(const Frag& A, const Frag& B, v8f C) {
    return __builtin_amdgcn_wmma_f32_16x16x32_bf16(false, A.v, false, B.v,
                                                   (short)0, C, false, false);
}

#define Bb   4
#define SEQ  4096
#define CDIM 128
#define NH   4
#define HD   32
#define HID  512
#define TOK  (Bb*SEQ)          // 16384 tokens
#define L2E  1.44269504088896f
#define QSCALE 0.1767766952966369f   // 1/sqrt(32)

// ---------------- K0: transpose+convert weights to bf16: dst[n*K+k] = src[k*Nc+n]
__global__ void __launch_bounds__(256) k_wconv(const float* __restrict__ src,
                                               unsigned short* __restrict__ dst,
                                               int K, int Nc) {
    int i = blockIdx.x * 256 + threadIdx.x;
    if (i >= K * Nc) return;
    int n = i / K, k = i % K;
    dst[i] = f2bf(src[(long)k * Nc + n]);
}

// ---------------- K1: LN1 + QKV GEMM -> Q(scaled,bf16 [b,h,n,d]) K(bf16 [b,h,n,d]) V(bf16 [b,h,d,n])
__global__ void __launch_bounds__(128) k_ln_qkv(
    const float* __restrict__ x, const float* __restrict__ g1, const float* __restrict__ b1,
    const unsigned short* __restrict__ qkvT, const float* __restrict__ qkv_b,
    unsigned short* __restrict__ Qb, unsigned short* __restrict__ Kb,
    unsigned short* __restrict__ Vt)
{
    __shared__ alignas(16) unsigned short Abuf[64 * 136];   // 64 rows x 128 bf16, stride 136
    const int t = threadIdx.x;
    const long blk = blockIdx.x;

    {   // phase 1: layernorm 64 rows (2 threads per row)
        int r = t >> 1, half = t & 1;
        const float* xr = x + (blk * 64 + r) * CDIM + half * 64;
        float v[64], s = 0.f, s2 = 0.f;
#pragma unroll
        for (int i = 0; i < 64; ++i) { float f = xr[i]; v[i] = f; s += f; s2 += f * f; }
        s  += __shfl_xor(s, 1);
        s2 += __shfl_xor(s2, 1);
        float mu   = s * (1.f / 128.f);
        float var  = s2 * (1.f / 128.f) - mu * mu;
        float rstd = rsqrtf(var + 1e-5f);
        unsigned short* dst = &Abuf[r * 136 + half * 64];
        const float* gg = g1 + half * 64;
        const float* bb = b1 + half * 64;
#pragma unroll
        for (int i = 0; i < 64; ++i) dst[i] = f2bf((v[i] - mu) * rstd * gg[i] + bb[i]);
    }
    __syncthreads();

    // phase 2: wave w computes rows [w*16, w*16+16) x 384 cols with bf16 WMMA
    int w = t >> 5, l = t & 31, ln = l & 15, hsel = l >> 4;
    Frag a[4];
#pragma unroll
    for (int kk = 0; kk < 4; ++kk) {
        int row = w * 16 + ln;
        int c1  = kk * 32 + hsel * 16;
        a[kk].u[0] = *(const v4u*)&Abuf[row * 136 + c1];
        a[kk].u[1] = *(const v4u*)&Abuf[row * 136 + c1 + 8];
    }

    long rowb = blk * 64 + w * 16 + hsel * 8;     // first of the 8 rows this half-lane owns
    int  b_   = (int)(rowb >> 12);                // uniform per block (64-row tiles don't cross b)
    int  nnb  = (int)(rowb & 4095);

    // one 16x16 output tile: batch 8 b128 loads, then 4 WMMAs on 2 independent chains
    auto tile = [&](int ct) -> v8f {
        int col = ct * 16 + ln;
        Frag bf[4];
#pragma unroll
        for (int kk = 0; kk < 4; ++kk) {
            int kb = kk * 32 + hsel * 16;
            bf[kk].u[0] = *(const v4u*)&qkvT[col * CDIM + kb];
            bf[kk].u[1] = *(const v4u*)&qkvT[col * CDIM + kb + 8];
        }
        v8f c0 = {}, c1 = {};
        c0 = wmma_bf16(a[0], bf[0], c0);
        c1 = wmma_bf16(a[1], bf[1], c1);
        c0 = wmma_bf16(a[2], bf[2], c0);
        c1 = wmma_bf16(a[3], bf[3], c1);
        return c0 + c1;
    };

    for (int ct = 0; ct < 8; ++ct) {              // ---- Q (scaled) ----
        v8f c = tile(ct);
        float bias = qkv_b[ct * 16 + ln];
        int hh = (ct >> 1) & 3, d = (ct & 1) * 16 + ln;       // scalar hh, uniform tile
        unsigned short* qp = Qb + ((long)(b_ * NH + hh) * SEQ + nnb) * HD + d;
#pragma unroll
        for (int g = 0; g < 8; ++g) qp[g * HD] = f2bf((c[g] + bias) * QSCALE);
    }
    for (int ct = 8; ct < 16; ++ct) {             // ---- K ----
        v8f c = tile(ct);
        float bias = qkv_b[ct * 16 + ln];
        int hh = (ct >> 1) & 3, d = (ct & 1) * 16 + ln;
        unsigned short* kp = Kb + ((long)(b_ * NH + hh) * SEQ + nnb) * HD + d;
#pragma unroll
        for (int g = 0; g < 8; ++g) kp[g * HD] = f2bf(c[g] + bias);
    }
    for (int ct = 16; ct < 24; ++ct) {            // ---- V (transposed: [d][n]) ----
        v8f c = tile(ct);
        float bias = qkv_b[ct * 16 + ln];
        int hh = (ct >> 1) & 3, d = (ct & 1) * 16 + ln;
        Pack8 st;
#pragma unroll
        for (int g = 0; g < 8; ++g) st.s[g] = f2bf(c[g] + bias);
        // tokens contiguous per lane -> single b128 store (nnb is 8-aligned)
        *(v4u*)&Vt[((long)(b_ * NH + hh) * HD + d) * SEQ + nnb] = st.u;
    }
}

// ---------------- K2: flash attention; block = (b,h,64 query rows); wave = 16 rows
__global__ void __launch_bounds__(128) k_attn(
    const unsigned short* __restrict__ Qb, const unsigned short* __restrict__ Kb,
    const unsigned short* __restrict__ Vt, unsigned short* __restrict__ attnout)
{
    __shared__ alignas(16) unsigned short Pbuf[4 * 16 * 72];   // per-wave 16x64 P tile, stride 72
    int t = threadIdx.x, w = t >> 5, l = t & 31, ln = l & 15, hsel = l >> 4;
    int bh = blockIdx.x >> 6, qblk = blockIdx.x & 63;
    int b_ = bh >> 2, h = bh & 3;
    long kvbase = (long)bh * SEQ * HD;
    int  n0 = qblk * 64 + w * 16;

    Frag qf;
    {
        const unsigned short* qp = Qb + kvbase + (long)(n0 + ln) * HD;
        int c1 = hsel * 16;
        qf.u[0] = *(const v4u*)&qp[c1];
        qf.u[1] = *(const v4u*)&qp[c1 + 8];
    }
    v8f o0 = {}, o1 = {};
    float mrow[8], lsum[8];
#pragma unroll
    for (int g = 0; g < 8; ++g) { mrow[g] = -1e30f; lsum[g] = 0.f; }
    unsigned short* Pw = &Pbuf[w * 16 * 72];

    auto loadK = [&](Frag* kf, int kc2) {
        int base = kc2 * 64;
#pragma unroll
        for (int tt = 0; tt < 4; ++tt) {
            const unsigned short* kp = Kb + kvbase + (long)(base + tt * 16 + ln) * HD + hsel * 16;
            kf[tt].u[0] = *(const v4u*)&kp[0];
            kf[tt].u[1] = *(const v4u*)&kp[8];
        }
    };

    Frag kf[4];
    loadK(kf, 0);                                  // prime the pipeline

    for (int kc = 0; kc < 64; ++kc) {
        int kb0 = kc * 64;
        v8f s[4];
#pragma unroll
        for (int tt = 0; tt < 4; ++tt) {           // scores: Q(16x32) x K^T(32x16), 4 indep chains
            v8f z = {};
            s[tt] = wmma_bf16(qf, kf[tt], z);
        }
        // issue V loads now -- they are consumed only after the softmax section
        Frag vf[4];                                // [kk*2 + nt]
#pragma unroll
        for (int kk = 0; kk < 2; ++kk) {
            int keyb = kb0 + kk * 32 + hsel * 16;
            const unsigned short* vp0 = Vt + kvbase + (long)(ln) * SEQ + keyb;
            const unsigned short* vp1 = Vt + kvbase + (long)(16 + ln) * SEQ + keyb;
            vf[kk * 2 + 0].u[0] = *(const v4u*)&vp0[0]; vf[kk * 2 + 0].u[1] = *(const v4u*)&vp0[8];
            vf[kk * 2 + 1].u[0] = *(const v4u*)&vp1[0]; vf[kk * 2 + 1].u[1] = *(const v4u*)&vp1[8];
        }
        if (kc + 1 < 64) loadK(kf, kc + 1);        // prefetch next chunk's K under softmax

#pragma unroll
        for (int g = 0; g < 8; ++g) {              // online softmax per row
            float tm = fmaxf(fmaxf(s[0][g], s[1][g]), fmaxf(s[2][g], s[3][g]));
            tm = fmaxf(tm, __shfl_xor(tm, 1));
            tm = fmaxf(tm, __shfl_xor(tm, 2));
            tm = fmaxf(tm, __shfl_xor(tm, 4));
            tm = fmaxf(tm, __shfl_xor(tm, 8));
            float nm = fmaxf(mrow[g], tm);
            float sc = exp2f((mrow[g] - nm) * L2E);
            mrow[g] = nm;
            o0[g] *= sc; o1[g] *= sc;
            float rs = 0.f;
            int prow = g + hsel * 8;
#pragma unroll
            for (int tt = 0; tt < 4; ++tt) {
                float p = exp2f((s[tt][g] - nm) * L2E);
                rs += p;
                Pw[prow * 72 + tt * 16 + ln] = f2bf(p);
            }
            rs += __shfl_xor(rs, 1); rs += __shfl_xor(rs, 2);
            rs += __shfl_xor(rs, 4); rs += __shfl_xor(rs, 8);
            lsum[g] = lsum[g] * sc + rs;
        }
        asm volatile("" ::: "memory");             // DS in-order per wave; stop compiler reordering
        Frag pa[2];
#pragma unroll
        for (int kk = 0; kk < 2; ++kk) {           // reload P as A fragments
            int c1 = kk * 32 + hsel * 16;
            pa[kk].u[0] = *(const v4u*)&Pw[ln * 72 + c1];
            pa[kk].u[1] = *(const v4u*)&Pw[ln * 72 + c1 + 8];
        }
        o0 = wmma_bf16(pa[0], vf[0], o0);          // 2 interleaved accumulation chains
        o1 = wmma_bf16(pa[0], vf[1], o1);
        o0 = wmma_bf16(pa[1], vf[2], o0);
        o1 = wmma_bf16(pa[1], vf[3], o1);
        asm volatile("" ::: "memory");
    }
#pragma unroll
    for (int g = 0; g < 8; ++g) {
        float inv = 1.f / lsum[g];
        long token = (long)b_ * SEQ + n0 + g + hsel * 8;
        attnout[token * CDIM + h * HD + ln]      = f2bf(o0[g] * inv);
        attnout[token * CDIM + h * HD + 16 + ln] = f2bf(o1[g] * inv);
    }
}

// ---------------- K3: proj GEMM + bias + residual -> x1 (f32), then LN2 -> hln (bf16)
__global__ void __launch_bounds__(128) k_proj_ln2(
    const unsigned short* __restrict__ attnout, const unsigned short* __restrict__ projT,
    const float* __restrict__ proj_b, const float* __restrict__ x,
    const float* __restrict__ g2, const float* __restrict__ b2,
    float* __restrict__ x1, unsigned short* __restrict__ hln)
{
    int t = threadIdx.x, w = t >> 5, l = t & 31, ln = l & 15, hsel = l >> 4;
    long rowbase = (long)blockIdx.x * 64 + w * 16;
    Frag a[4];
#pragma unroll
    for (int kk = 0; kk < 4; ++kk) {
        long row = rowbase + ln;
        int  c1  = kk * 32 + hsel * 16;
        a[kk].u[0] = *(const v4u*)&attnout[row * CDIM + c1];
        a[kk].u[1] = *(const v4u*)&attnout[row * CDIM + c1 + 8];
    }
    float s1[8] = {}, s2v[8] = {};
    for (int ct = 0; ct < 8; ++ct) {
        int col = ct * 16 + ln;
        Frag bf[4];
#pragma unroll
        for (int kk = 0; kk < 4; ++kk) {
            int kb = kk * 32 + hsel * 16;
            bf[kk].u[0] = *(const v4u*)&projT[col * CDIM + kb];
            bf[kk].u[1] = *(const v4u*)&projT[col * CDIM + kb + 8];
        }
        v8f c0 = {}, c1 = {};
        c0 = wmma_bf16(a[0], bf[0], c0);
        c1 = wmma_bf16(a[1], bf[1], c1);
        c0 = wmma_bf16(a[2], bf[2], c0);
        c1 = wmma_bf16(a[3], bf[3], c1);
        v8f c = c0 + c1;
        float bias = proj_b[col];
#pragma unroll
        for (int g = 0; g < 8; ++g) {
            long token = rowbase + g + hsel * 8;
            float val = c[g] + bias + x[token * CDIM + col];
            x1[token * CDIM + col] = val;
            s1[g] += val; s2v[g] += val * val;
        }
    }
#pragma unroll
    for (int g = 0; g < 8; ++g) {                   // LN2 stats (full row = 16 lanes x 8 cols)
        float a1 = s1[g], a2 = s2v[g];
        a1 += __shfl_xor(a1, 1); a2 += __shfl_xor(a2, 1);
        a1 += __shfl_xor(a1, 2); a2 += __shfl_xor(a2, 2);
        a1 += __shfl_xor(a1, 4); a2 += __shfl_xor(a2, 4);
        a1 += __shfl_xor(a1, 8); a2 += __shfl_xor(a2, 8);
        float mu  = a1 * (1.f / 128.f);
        float var = a2 * (1.f / 128.f) - mu * mu;
        s1[g]  = mu;
        s2v[g] = rsqrtf(var + 1e-5f);
    }
    asm volatile("" ::: "memory");
    for (int ct = 0; ct < 8; ++ct) {
        int col = ct * 16 + ln;
        float gg = g2[col], bb = b2[col];
#pragma unroll
        for (int g = 0; g < 8; ++g) {
            long token = rowbase + g + hsel * 8;
            float v = x1[token * CDIM + col];
            hln[token * CDIM + col] = f2bf((v - s1[g]) * s2v[g] * gg + bb);
        }
    }
}

// ---------------- K4a: MLP1 (128->512) + exact GELU -> hidden (bf16)
__global__ void __launch_bounds__(128) k_mlp1(
    const unsigned short* __restrict__ hln, const unsigned short* __restrict__ w1T,
    const float* __restrict__ b1, unsigned short* __restrict__ hidden)
{
    int t = threadIdx.x, w = t >> 5, l = t & 31, ln = l & 15, hsel = l >> 4;
    long rowbase = (long)blockIdx.x * 64 + w * 16;
    Frag a[4];
#pragma unroll
    for (int kk = 0; kk < 4; ++kk) {
        long row = rowbase + ln;
        int  c1  = kk * 32 + hsel * 16;
        a[kk].u[0] = *(const v4u*)&hln[row * CDIM + c1];
        a[kk].u[1] = *(const v4u*)&hln[row * CDIM + c1 + 8];
    }
    for (int ct = 0; ct < 32; ++ct) {
        int col = ct * 16 + ln;
        Frag bf[4];
#pragma unroll
        for (int kk = 0; kk < 4; ++kk) {
            int kb = kk * 32 + hsel * 16;
            bf[kk].u[0] = *(const v4u*)&w1T[(long)col * CDIM + kb];
            bf[kk].u[1] = *(const v4u*)&w1T[(long)col * CDIM + kb + 8];
        }
        v8f c0 = {}, c1 = {};
        c0 = wmma_bf16(a[0], bf[0], c0);
        c1 = wmma_bf16(a[1], bf[1], c1);
        c0 = wmma_bf16(a[2], bf[2], c0);
        c1 = wmma_bf16(a[3], bf[3], c1);
        v8f c = c0 + c1;
        float bias = b1[col];
#pragma unroll
        for (int g = 0; g < 8; ++g) {
            long token = rowbase + g + hsel * 8;
            float v  = c[g] + bias;
            float ge = 0.5f * v * (1.f + erff(v * 0.70710678118f));   // exact GELU
            hidden[token * HID + col] = f2bf(ge);
        }
    }
}

// ---------------- K4b: MLP2 (512->128) + bias + residual -> d_out (f32)
__global__ void __launch_bounds__(128) k_mlp2(
    const unsigned short* __restrict__ hidden, const unsigned short* __restrict__ w2T,
    const float* __restrict__ b2, const float* __restrict__ x1, float* __restrict__ out)
{
    int t = threadIdx.x, w = t >> 5, l = t & 31, ln = l & 15, hsel = l >> 4;
    long rowbase = (long)blockIdx.x * 64 + w * 16;
    v8f c[8];
#pragma unroll
    for (int ct = 0; ct < 8; ++ct) c[ct] = (v8f){};
    for (int kk = 0; kk < 16; ++kk) {
        Frag a;
        long row = rowbase + ln;
        int  c1  = kk * 32 + hsel * 16;
        a.u[0] = *(const v4u*)&hidden[row * HID + c1];
        a.u[1] = *(const v4u*)&hidden[row * HID + c1 + 8];
        Frag bf[8];
#pragma unroll
        for (int ct = 0; ct < 8; ++ct) {
            int col = ct * 16 + ln;
            int kb  = kk * 32 + hsel * 16;
            bf[ct].u[0] = *(const v4u*)&w2T[(long)col * HID + kb];
            bf[ct].u[1] = *(const v4u*)&w2T[(long)col * HID + kb + 8];
        }
#pragma unroll
        for (int ct = 0; ct < 8; ++ct) c[ct] = wmma_bf16(a, bf[ct], c[ct]);   // 8 indep chains
    }
#pragma unroll
    for (int ct = 0; ct < 8; ++ct) {
        int col = ct * 16 + ln;
        float bias = b2[col];
#pragma unroll
        for (int g = 0; g < 8; ++g) {
            long token = rowbase + g + hsel * 8;
            out[token * CDIM + col] = c[ct][g] + bias + x1[token * CDIM + col];
        }
    }
}

// ---------------- launch ----------------
extern "C" void kernel_launch(void* const* d_in, const int* in_sizes, int n_in,
                              void* d_out, int out_size, void* d_ws, size_t ws_size,
                              hipStream_t stream) {
    const float* x      = (const float*)d_in[0];
    const float* ln1_g  = (const float*)d_in[1];
    const float* ln1_b  = (const float*)d_in[2];
    const float* qkv_w  = (const float*)d_in[3];
    const float* qkv_b  = (const float*)d_in[4];
    const float* proj_w = (const float*)d_in[5];
    const float* proj_b = (const float*)d_in[6];
    const float* ln2_g  = (const float*)d_in[7];
    const float* ln2_b  = (const float*)d_in[8];
    const float* mlp_w1 = (const float*)d_in[9];
    const float* mlp_b1 = (const float*)d_in[10];
    const float* mlp_w2 = (const float*)d_in[11];
    const float* mlp_b2 = (const float*)d_in[12];

    char* ws = (char*)d_ws;
    unsigned short* qkvT   = (unsigned short*)(ws);              //  98304 B
    unsigned short* projT  = (unsigned short*)(ws + 98304);      //  32768 B
    unsigned short* w1T    = (unsigned short*)(ws + 131072);     // 131072 B
    unsigned short* w2T    = (unsigned short*)(ws + 262144);     // 131072 B
    unsigned short* Qb     = (unsigned short*)(ws + 393216);     // 4 MiB
    unsigned short* Kb     = (unsigned short*)(ws + 4587520);    // 4 MiB
    unsigned short* Vt     = (unsigned short*)(ws + 8781824);    // 4 MiB
    unsigned short* attn   = (unsigned short*)(ws + 12976128);   // 4 MiB
    float*          x1     = (float*)(ws + 17170432);            // 8 MiB
    unsigned short* hln    = (unsigned short*)(ws + 25559040);   // 4 MiB
    unsigned short* hidden = (unsigned short*)(ws + 29753344);   // 16 MiB  (end 46530560)

    k_wconv<<<192, 256, 0, stream>>>(qkv_w,  qkvT, 128, 384);
    k_wconv<<<64,  256, 0, stream>>>(proj_w, projT, 128, 128);
    k_wconv<<<256, 256, 0, stream>>>(mlp_w1, w1T,  128, 512);
    k_wconv<<<256, 256, 0, stream>>>(mlp_w2, w2T,  512, 128);

    k_ln_qkv  <<<TOK / 64, 128, 0, stream>>>(x, ln1_g, ln1_b, qkvT, qkv_b, Qb, Kb, Vt);
    k_attn    <<<Bb * NH * (SEQ / 64), 128, 0, stream>>>(Qb, Kb, Vt, attn);
    k_proj_ln2<<<TOK / 64, 128, 0, stream>>>(attn, projT, proj_b, x, ln2_g, ln2_b, x1, hln);
    k_mlp1    <<<TOK / 64, 128, 0, stream>>>(hln, w1T, mlp_b1, hidden);
    k_mlp2    <<<TOK / 64, 128, 0, stream>>>(hidden, w2T, mlp_b2, x1, (float*)d_out);
}
